// CADTopoEncoder_68281390071947
// MI455X (gfx1250) — compile-verified, matrix operand order checked by reading, not run on hardware
//
#include <hip/hip_runtime.h>

#define HID 64
#define NPTS 200000
#define NEDG 100000
#define NFAC 50000
#define EDG  500000
#define NB   64

typedef __attribute__((ext_vector_type(16))) __bf16 v16bf;
typedef __attribute__((ext_vector_type(8)))  __bf16 v8bf;
typedef __attribute__((ext_vector_type(8)))  float  v8f;
typedef __attribute__((ext_vector_type(4)))  float  f32x4;

// ---------------------------------------------------------------- utilities
__global__ void fill_zero_kernel(float* __restrict__ p, long n) {
  long i = (long)blockIdx.x * blockDim.x + threadIdx.x;
  long st = (long)gridDim.x * blockDim.x;
  for (; i < n; i += st) p[i] = 0.0f;
}

// ------------------------------------------------- first MLP layer (tiny K)
// t[node,j] = relu(b1[j] + sum_k x[node,k] * w1[k,j])
__global__ void mlp1_kernel(const float* __restrict__ x, const float* __restrict__ w1,
                            const float* __restrict__ b1, float* __restrict__ t,
                            int n, int in_f) {
  long i  = (long)blockIdx.x * blockDim.x + threadIdx.x;
  long st = (long)gridDim.x * blockDim.x;
  long total = (long)n * HID;
  for (; i < total; i += st) {
    long node = i >> 6; int j = (int)(i & 63);
    float acc = b1[j];
    const float* xr = x + node * in_f;
    for (int k = 0; k < in_f; ++k) acc += xr[k] * w1[k * HID + j];
    t[i] = fmaxf(acc, 0.0f);
  }
}

// --------------------------------------------------------- WMMA 64x64 GEMM
// C[r,:] (+)= (A[r,:] * scale_r) @ W + bias ;  scale_r = 1/max(cnt[r],1)
// REQUIRES n % 16 == 0 (true for all call sites: 200000/100000/50000).
// 256 threads = 8 waves, each wave owns a 16-row tile -> 128 rows per block.
// W staged in LDS as bf16, transposed, stride 72 (16B aligned, conflict-free
// starting banks: 36*col mod 64 distinct multiples of 4 for col 0..15).
#define WSTR 72
template <int ACCUM>
__global__ __launch_bounds__(256)
void gemm64_wmma_kernel(const float* __restrict__ A, const float* __restrict__ cnt,
                        const float* __restrict__ W, const float* __restrict__ bias,
                        float* __restrict__ C, int n) {
  __shared__ __attribute__((aligned(16))) __bf16 sW[HID * WSTR]; // sW[col*WSTR+k]
  __shared__ float sB[HID];
  int tid = threadIdx.x;
  for (int idx = tid; idx < HID * HID; idx += 256) {
    int k = idx >> 6, col = idx & 63;
    sW[col * WSTR + k] = (__bf16)W[idx];
  }
  if (tid < HID) sB[tid] = bias ? bias[tid] : 0.0f;
  __syncthreads();

  int wave = tid >> 5, lane = tid & 31;
  int m = lane & 15, half = lane >> 4;
  long rowBase = (long)blockIdx.x * 128 + (long)wave * 16;
  if (rowBase >= n) return;               // uniform per wave; tiles are full

  // ---- A fragments for both k-steps (ISA 16-bit A 16x32 layout),
  //      loaded as float4 (global_load_b128) and packed to bf16.
  long r = rowBase + m;
  float scale = cnt ? (1.0f / fmaxf(cnt[r], 1.0f)) : 1.0f;
  const f32x4* a4 = (const f32x4*)(A + r * HID);
  v16bf afrag[2];
#pragma unroll
  for (int ks = 0; ks < 2; ++ks) {
    int q = (ks * 32 + half * 8) >> 2;    // float4 index; k0 multiple of 8
    f32x4 x0 = a4[q], x1 = a4[q + 1];     // K = k0 .. k0+7
    f32x4 x2 = a4[q + 4], x3 = a4[q + 5]; // K = k0+16 .. k0+23
#pragma unroll
    for (int e = 0; e < 4; ++e) {
      afrag[ks][e]      = (__bf16)(x0[e] * scale);
      afrag[ks][e + 4]  = (__bf16)(x1[e] * scale);
      afrag[ks][e + 8]  = (__bf16)(x2[e] * scale);
      afrag[ks][e + 12] = (__bf16)(x3[e] * scale);
    }
  }

  // ---- accumulators: 4 N-tiles, col = t*16 + (lane&15); M = e + 8*half
  int col0 = m;                            // tile t adds t*16
  v8f acc[4];
#pragma unroll
  for (int t = 0; t < 4; ++t) {
    int col = t * 16 + col0;
    float bv = sB[col];
#pragma unroll
    for (int e = 0; e < 8; ++e) {
      long rr = rowBase + e + 8 * half;
      acc[t][e] = (ACCUM ? C[rr * HID + col] : 0.0f) + bv;
    }
  }

  // ---- ks outer / tile inner: 4 independent WMMAs in flight per k-step
#pragma unroll
  for (int ks = 0; ks < 2; ++ks) {
    int kb = ks * 32 + half * 8;
#pragma unroll
    for (int t = 0; t < 4; ++t) {
      int col = t * 16 + col0;
      const v8bf* wp = (const v8bf*)(&sW[col * WSTR + kb]); // 16B aligned
      v8bf lo = wp[0];          // K = kb .. kb+7
      v8bf hi = wp[2];          // K = kb+16 .. kb+23
      v16bf bfrag;
#pragma unroll
      for (int e = 0; e < 8; ++e) { bfrag[e] = lo[e]; bfrag[e + 8] = hi[e]; }
      acc[t] = __builtin_amdgcn_wmma_f32_16x16x32_bf16(
          false, afrag[ks], false, bfrag, (short)0, acc[t], false, false);
    }
  }

  // ---- straight-line stores (no guards: n % 16 == 0)
#pragma unroll
  for (int t = 0; t < 4; ++t) {
    int col = t * 16 + col0;
#pragma unroll
    for (int e = 0; e < 8; ++e) {
      long rr = rowBase + e + 8 * half;
      C[rr * HID + col] = acc[t][e];
    }
  }
}

// ------------------------------------------------------------ edge scatter
__global__ void scatter_feat_kernel(const float* __restrict__ hs,
                                    const int* __restrict__ src,
                                    const int* __restrict__ dst,
                                    float* __restrict__ s, int ne) {
  long i  = (long)blockIdx.x * blockDim.x + threadIdx.x;
  long st = (long)gridDim.x * blockDim.x;
  long total = (long)ne * HID;
  for (; i < total; i += st) {
    long e = i >> 6; int f = (int)(i & 63);
    long eAhead = (i + st) >> 6;
    if (eAhead < ne)  // global_prefetch for the next grid-stride iteration
      __builtin_prefetch(&hs[(long)src[eAhead] * HID + f], 0, 0);
    atomicAdd(&s[(long)dst[e] * HID + f], hs[(long)src[e] * HID + f]);
  }
}

__global__ void scatter_cnt_kernel(const int* __restrict__ dst,
                                   float* __restrict__ c, int ne) {
  long i  = (long)blockIdx.x * blockDim.x + threadIdx.x;
  long st = (long)gridDim.x * blockDim.x;
  for (; i < ne; i += st) atomicAdd(&c[dst[i]], 1.0f);
}

// ------------------------------------- h = LN(h + relu(acc)) * g + b  (wave/row)
__global__ __launch_bounds__(256)
void ln_update_kernel(float* __restrict__ h, const float* __restrict__ acc,
                      const float* __restrict__ g, const float* __restrict__ b,
                      int n) {
  int wave = threadIdx.x >> 5, lane = threadIdx.x & 31;
  long row = (long)blockIdx.x * 8 + wave;
  if (row >= n) return;
  long base = row * HID;
  float v0 = h[base + lane]      + fmaxf(acc[base + lane], 0.0f);
  float v1 = h[base + lane + 32] + fmaxf(acc[base + lane + 32], 0.0f);
  float s = v0 + v1;
  for (int off = 16; off > 0; off >>= 1) s += __shfl_xor(s, off, 32);
  float mean = s * (1.0f / 64.0f);
  float d0 = v0 - mean, d1 = v1 - mean;
  float vv = d0 * d0 + d1 * d1;
  for (int off = 16; off > 0; off >>= 1) vv += __shfl_xor(vv, off, 32);
  float rstd = rsqrtf(vv * (1.0f / 64.0f) + 1e-5f);
  h[base + lane]      = d0 * rstd * g[lane]      + b[lane];
  h[base + lane + 32] = d1 * rstd * g[lane + 32] + b[lane + 32];
}

// ----------------------------------------------------------------- pooling
__global__ void pool_scatter_kernel(const float* __restrict__ h,
                                    const int* __restrict__ batch,
                                    float* __restrict__ psum,
                                    float* __restrict__ pcnt,  // per-type, 64 slots
                                    int n, int colOff) {
  long i  = (long)blockIdx.x * blockDim.x + threadIdx.x;
  long st = (long)gridDim.x * blockDim.x;
  long total = (long)n * HID;
  for (; i < total; i += st) {
    long node = i >> 6; int f = (int)(i & 63);
    int bb = batch[node];
    atomicAdd(&psum[(long)bb * 192 + colOff + f], h[i]);
    if (f == 0) atomicAdd(&pcnt[bb], 1.0f);
  }
}

__global__ void pool_finalize_kernel(const float* __restrict__ psum,
                                     const float* __restrict__ pcnt,
                                     float* __restrict__ out) {
  int i = blockIdx.x * blockDim.x + threadIdx.x;
  if (i >= NB * 192) return;
  int bb = i / 192, col = i % 192, type = col / 64;
  out[i] = psum[i] / fmaxf(pcnt[type * 64 + bb], 1.0f);
}

// ------------------------------------------------------------------- host
static inline int cdiv(long a, long b) { return (int)((a + b - 1) / b); }
static inline int capg(long work, int cap) { int g = cdiv(work, 256); return g < cap ? (g < 1 ? 1 : g) : cap; }

extern "C" void kernel_launch(void* const* d_in, const int* in_sizes, int n_in,
                              void* d_out, int out_size, void* d_ws, size_t ws_size,
                              hipStream_t stream) {
  const float* point_x = (const float*)d_in[0];
  const float* edge_x  = (const float*)d_in[1];
  const float* face_x  = (const float*)d_in[2];
  const int* batch_point = (const int*)d_in[3];
  const int* batch_edge  = (const int*)d_in[4];
  const int* batch_face  = (const int*)d_in[5];
  const int* ei[8];
  for (int r = 0; r < 8; ++r) ei[r] = (const int*)d_in[6 + r];
  const float* pw1 = (const float*)d_in[14]; const float* pb1 = (const float*)d_in[15];
  const float* pw2 = (const float*)d_in[16]; const float* pb2 = (const float*)d_in[17];
  const float* ew1 = (const float*)d_in[18]; const float* eb1 = (const float*)d_in[19];
  const float* ew2 = (const float*)d_in[20]; const float* eb2 = (const float*)d_in[21];
  const float* fw1 = (const float*)d_in[22]; const float* fb1 = (const float*)d_in[23];
  const float* fw2 = (const float*)d_in[24]; const float* fb2 = (const float*)d_in[25];
  const float* Wl  = (const float*)d_in[26];
  const float* bl  = (const float*)d_in[27];
  const float* Wr  = (const float*)d_in[28];
  const float* ln_g = (const float*)d_in[29];
  const float* ln_b = (const float*)d_in[30];
  float* out = (float*)d_out;

  // workspace layout (floats)
  float* ws = (float*)d_ws;
  float* hp   = ws;               ws += (long)NPTS * HID;
  float* he   = ws;               ws += (long)NEDG * HID;
  float* hf   = ws;               ws += (long)NFAC * HID;
  float* accP = ws;               ws += (long)NPTS * HID;
  float* accE = ws;               ws += (long)NEDG * HID;
  float* accF = ws;               ws += (long)NFAC * HID;
  float* sbuf = ws;               ws += (long)NPTS * HID;   // max dst rows
  float* cbuf = ws;               ws += (long)NPTS;
  float* psum = ws;               ws += (long)NB * 192;
  float* pcnt = ws;               ws += 3 * 64;

  // ---- encoders: h = relu(x@w1+b1)@w2 + b2
  mlp1_kernel<<<capg((long)NPTS * HID, 4096), 256, 0, stream>>>(point_x, pw1, pb1, sbuf, NPTS, 3);
  gemm64_wmma_kernel<0><<<cdiv(NPTS, 128), 256, 0, stream>>>(sbuf, nullptr, pw2, pb2, hp, NPTS);
  mlp1_kernel<<<capg((long)NEDG * HID, 4096), 256, 0, stream>>>(edge_x, ew1, eb1, sbuf, NEDG, 2);
  gemm64_wmma_kernel<0><<<cdiv(NEDG, 128), 256, 0, stream>>>(sbuf, nullptr, ew2, eb2, he, NEDG);
  mlp1_kernel<<<capg((long)NFAC * HID, 4096), 256, 0, stream>>>(face_x, fw1, fb1, sbuf, NFAC, 16);
  gemm64_wmma_kernel<0><<<cdiv(NFAC, 128), 256, 0, stream>>>(sbuf, nullptr, fw2, fb2, hf, NFAC);

  // relation tables: pp, fp, ep, pf, ef, ff, pe, fe
  const float* srcH[8] = { hp, hf, he, hp, he, hf, hp, hf };
  const float* dstH[8] = { hp, hp, hp, hf, hf, hf, he, he };
  int          ndst[8] = { NPTS, NPTS, NPTS, NFAC, NFAC, NFAC, NEDG, NEDG };
  float*       accT[8] = { accP, accP, accP, accF, accF, accF, accE, accE };

  for (int layer = 0; layer < 2; ++layer) {
    fill_zero_kernel<<<4096, 256, 0, stream>>>(accP, (long)NPTS * HID);
    fill_zero_kernel<<<2048, 256, 0, stream>>>(accE, (long)NEDG * HID);
    fill_zero_kernel<<<1024, 256, 0, stream>>>(accF, (long)NFAC * HID);
    for (int r = 0; r < 8; ++r) {
      int nd = ndst[r];
      const int* src_idx = ei[r];
      const int* dst_idx = ei[r] + EDG;
      fill_zero_kernel<<<capg((long)nd * HID, 4096), 256, 0, stream>>>(sbuf, (long)nd * HID);
      fill_zero_kernel<<<capg(nd, 1024), 256, 0, stream>>>(cbuf, nd);
      scatter_feat_kernel<<<4096, 256, 0, stream>>>(srcH[r], src_idx, dst_idx, sbuf, EDG);
      scatter_cnt_kernel<<<2048, 256, 0, stream>>>(dst_idx, cbuf, EDG);
      const float* wl = Wl + (long)(layer * 8 + r) * HID * HID;
      const float* bb = bl + (long)(layer * 8 + r) * HID;
      const float* wr = Wr + (long)(layer * 8 + r) * HID * HID;
      gemm64_wmma_kernel<1><<<cdiv(nd, 128), 256, 0, stream>>>(sbuf, cbuf, wl, bb, accT[r], nd);
      gemm64_wmma_kernel<1><<<cdiv(nd, 128), 256, 0, stream>>>(dstH[r], nullptr, wr, nullptr, accT[r], nd);
    }
    ln_update_kernel<<<cdiv(NPTS, 8), 256, 0, stream>>>(hp, accP, ln_g + (long)(layer * 3 + 0) * HID, ln_b + (long)(layer * 3 + 0) * HID, NPTS);
    ln_update_kernel<<<cdiv(NEDG, 8), 256, 0, stream>>>(he, accE, ln_g + (long)(layer * 3 + 1) * HID, ln_b + (long)(layer * 3 + 1) * HID, NEDG);
    ln_update_kernel<<<cdiv(NFAC, 8), 256, 0, stream>>>(hf, accF, ln_g + (long)(layer * 3 + 2) * HID, ln_b + (long)(layer * 3 + 2) * HID, NFAC);
  }

  // ---- pooling + concat
  fill_zero_kernel<<<16, 256, 0, stream>>>(psum, (long)NB * 192);
  fill_zero_kernel<<<1, 256, 0, stream>>>(pcnt, 3 * 64);
  pool_scatter_kernel<<<4096, 256, 0, stream>>>(hp, batch_point, psum, pcnt + 0,   NPTS, 0);
  pool_scatter_kernel<<<2048, 256, 0, stream>>>(he, batch_edge,  psum, pcnt + 64,  NEDG, 64);
  pool_scatter_kernel<<<1024, 256, 0, stream>>>(hf, batch_face,  psum, pcnt + 128, NFAC, 128);
  pool_finalize_kernel<<<cdiv(NB * 192, 256), 256, 0, stream>>>(psum, pcnt, out);
}
